// Sim_GCN_12137577579121
// MI455X (gfx1250) — matrix operand order repaired and checked by reading, not compiled, not run on hardware
//
#include <hip/hip_runtime.h>

typedef __attribute__((ext_vector_type(16))) __bf16 v16bf;
typedef __attribute__((ext_vector_type(8)))  float  v8f;
typedef __attribute__((ext_vector_type(4)))  int    v4i;
typedef unsigned long long ull;
typedef unsigned short u16;

#define BATCH 64
#define NTOK  512
#define DIM   256
#define IMGT  51

#if __has_builtin(__builtin_amdgcn_global_load_async_to_lds_b128) && \
    __has_builtin(__builtin_amdgcn_s_wait_asynccnt)
#define USE_ASYNC_LDS 1
#else
#define USE_ASYNC_LDS 0
#endif

typedef __attribute__((address_space(1))) v4i gv4i;   // global
typedef __attribute__((address_space(3))) v4i lv4i;   // LDS

// 16B global -> LDS copy: async-DMA on gfx1250 toolchains that expose it, VGPR bounce otherwise.
__device__ __forceinline__ void cp16_g2l(u16* lds_dst, const u16* g_src) {
#if USE_ASYNC_LDS
  __builtin_amdgcn_global_load_async_to_lds_b128((gv4i*)g_src, (lv4i*)lds_dst, 0, 0);
#else
  *(uint4*)lds_dst = *(const uint4*)g_src;
#endif
}

__device__ __forceinline__ void async_fence() {
#if USE_ASYNC_LDS
  __builtin_amdgcn_s_wait_asynccnt(0);
#endif
}

__device__ __forceinline__ u16 f2bf(float f) {
  union { float f; unsigned u; } v; v.f = f;
  unsigned u = v.u;
  u += 0x7FFFu + ((u >> 16) & 1u);        // round-to-nearest-even
  return (u16)(u >> 16);
}

union FragU { v16bf v; unsigned u[8]; };

// ---------------- f32 -> bf16 bulk convert ----------------
__global__ __launch_bounds__(256) void cvt_bf16_kernel(const float* __restrict__ in,
                                                       u16* __restrict__ out, size_t n) {
  for (size_t i = (size_t)blockIdx.x * 256 + threadIdx.x; i < n; i += (size_t)gridDim.x * 256)
    out[i] = f2bf(in[i]);
}

// ---------------- L2 normalize rows of [B*N, D], bf16 out ----------------
__global__ __launch_bounds__(256) void l2norm_kernel(const float* __restrict__ in,
                                                     u16* __restrict__ out) {
  size_t row = blockIdx.x;
  float x = in[row * DIM + threadIdx.x];
  float s = x * x;
  #pragma unroll
  for (int o = 16; o > 0; o >>= 1) s += __shfl_xor(s, o);
  __shared__ float parts[8];
  if ((threadIdx.x & 31) == 0) parts[threadIdx.x >> 5] = s;
  __syncthreads();
  float tot = 0.f;
  #pragma unroll
  for (int i = 0; i < 8; ++i) tot += parts[i];
  float nrm = fmaxf(sqrtf(tot), 1e-8f);
  out[row * DIM + threadIdx.x] = f2bf(x / nrm);
}

// img/text block masking for the Gram epilogue
__device__ __forceinline__ float apply_mask(float v, int r, int c, const int* nm) {
  if (r < IMGT) return (c < IMGT) ? 0.f : v * (float)nm[c];
  if (c < IMGT) return v * (float)nm[r];
  return v;
}

// ---------------- batched bf16 WMMA GEMM ----------------
// TRANS_B=1: C[m][n] = sum_k A[m][k]*B[n][k]   (B row-major [N][K] bf16)
// TRANS_B=0: C[m][n] = sum_k A[m][k]*B[k][n]   (B row-major [K][N] bf16)
// BIAS_RELU: C = relu(C + bias[n]);  MASK: Gram block/noun masking;  OUT_F32: f32 store else bf16
// Block tile BM=128 x BN=64, BK=32; 8 wave32s in a 4x2 grid, each wave owns 32x32 of C (4 WMMAs/K-step).
template <bool TRANS_B, bool BIAS_RELU, bool MASK, bool OUT_F32>
__global__ __launch_bounds__(256) void gemm_bf16_wmma(
    const u16* __restrict__ A, const u16* __restrict__ Bg,
    const float* __restrict__ bias, const int* __restrict__ noun,
    void* __restrict__ C, int K, int lda, int ldb, int ldc,
    long sA, long sB, long sC) {
  constexpr int BM = 128, BN = 64, BK = 32, LS = 56;  // LS*2 = 112B: 16B-aligned, bank-conflict-free
  __shared__ __align__(16) u16 alds[BM * LS];
  __shared__ __align__(16) u16 blds[BN * LS];

  int b = blockIdx.z;
  A += (long)b * sA; Bg += (long)b * sB;
  const int* nm = MASK ? (noun + b * NTOK) : nullptr;

  int m0 = blockIdx.y * BM, n0 = blockIdx.x * BN;
  int tid = threadIdx.x, lane = tid & 31, wave = tid >> 5;
  int wm = (wave >> 1) * 32;   // 0,32,64,96
  int wn = (wave & 1) * 32;    // 0,32
  int l15 = lane & 15, hi = lane >> 4;

  v8f acc00 = {}, acc01 = {}, acc10 = {}, acc11 = {};

  for (int k0 = 0; k0 < K; k0 += BK) {
    // Stage A tile: 128x32 bf16, 16B per lane (async DMA to LDS when available).
    #pragma unroll
    for (int it = 0; it < 2; ++it) {
      int i = tid + it * 256;            // 0..511
      int m = i >> 2, ko = (i & 3) * 8;
      cp16_g2l(&alds[m * LS + ko], A + (size_t)(m0 + m) * lda + k0 + ko);
    }
    // Stage B tile as [n][k].
    if (TRANS_B) {
      int n = tid >> 2, ko = (tid & 3) * 8;
      cp16_g2l(&blds[n * LS + ko], Bg + (size_t)(n0 + n) * ldb + k0 + ko);
    } else {
      #pragma unroll
      for (int it = 0; it < 2; ++it) {
        int i = tid + it * 256;          // 0..511
        int k = i >> 4, nn = (i & 15) * 4;
        union { ull q; u16 s[4]; } u;
        u.q = *(const ull*)(Bg + (size_t)(k0 + k) * ldb + n0 + nn);
        #pragma unroll
        for (int j = 0; j < 4; ++j) blds[(nn + j) * LS + k] = u.s[j];
      }
    }
    async_fence();
    __syncthreads();

    // A frags: lane = row M; low lanes K0..7 & 16..23, high lanes K8..15 & 24..31.
    FragU a0, a1, b0, b1;
    const unsigned* ar0 = (const unsigned*)&alds[(wm + l15) * LS];
    const unsigned* ar1 = (const unsigned*)&alds[(wm + 16 + l15) * LS];
    #pragma unroll
    for (int i = 0; i < 4; ++i) {
      a0.u[i] = ar0[hi * 4 + i];  a0.u[4 + i] = ar0[8 + hi * 4 + i];
      a1.u[i] = ar1[hi * 4 + i];  a1.u[4 + i] = ar1[8 + hi * 4 + i];
    }
    // B frags: lane = col N; low lanes K0..7 & 16..23, high lanes K8..15 & 24..31.
    const unsigned* br0 = (const unsigned*)&blds[(wn + l15) * LS];
    const unsigned* br1 = (const unsigned*)&blds[(wn + 16 + l15) * LS];
    #pragma unroll
    for (int i = 0; i < 8; ++i) { b0.u[i] = br0[hi * 4 + i + (i >> 2) * 4]; b1.u[i] = br1[hi * 4 + i + (i >> 2) * 4]; }

    acc00 = __builtin_amdgcn_wmma_f32_16x16x32_bf16(false, a0.v, false, b0.v, (short)0, acc00, false, false);
    acc01 = __builtin_amdgcn_wmma_f32_16x16x32_bf16(false, a0.v, false, b1.v, (short)0, acc01, false, false);
    acc10 = __builtin_amdgcn_wmma_f32_16x16x32_bf16(false, a1.v, false, b0.v, (short)0, acc10, false, false);
    acc11 = __builtin_amdgcn_wmma_f32_16x16x32_bf16(false, a1.v, false, b1.v, (short)0, acc11, false, false);
    __syncthreads();
  }

  // Epilogue. C layout per tile: VGPR r -> (M = r + hi*8, N = l15).
  int colBase = n0 + wn + l15;
  int rowBase = m0 + wm + hi * 8;
  float bv0 = BIAS_RELU ? bias[colBase] : 0.f;
  float bv1 = BIAS_RELU ? bias[colBase + 16] : 0.f;

  float* Cf  = (float*)C + (OUT_F32 ? (long)b * sC : 0);
  u16*   Cb  = (u16*)C   + (OUT_F32 ? 0 : (long)b * sC);

  #pragma unroll
  for (int sub = 0; sub < 4; ++sub) {
    const v8f& acc = (sub == 0) ? acc00 : (sub == 1) ? acc01 : (sub == 2) ? acc10 : acc11;
    int row0 = rowBase + ((sub >> 1) ? 16 : 0);
    int col  = colBase + ((sub & 1) ? 16 : 0);
    float bv = (sub & 1) ? bv1 : bv0;
    #pragma unroll
    for (int r = 0; r < 8; ++r) {
      float v = acc[r];
      int row = row0 + r;
      if (MASK)      v = apply_mask(v, row, col, nm);
      if (BIAS_RELU) v = fmaxf(v + bv, 0.f);
      size_t idx = (size_t)row * ldc + col;
      if (OUT_F32) Cf[idx] = v; else Cb[idx] = f2bf(v);
    }
  }
}

extern "C" void kernel_launch(void* const* d_in, const int* in_sizes, int n_in,
                              void* d_out, int out_size, void* d_ws, size_t ws_size,
                              hipStream_t stream) {
  const float* input_1 = (const float*)d_in[0];
  const float* input_2 = (const float*)d_in[1];
  const int*   noun    = (const int*)d_in[3];
  const float* w0 = (const float*)d_in[4];
  const float* b0 = (const float*)d_in[5];
  const float* w1 = (const float*)d_in[6];
  const float* b1 = (const float*)d_in[7];
  const float* wf = (const float*)d_in[8];
  const float* bf = (const float*)d_in[9];
  float* out = (float*)d_out;

  const size_t ND  = (size_t)BATCH * NTOK * DIM;     // 8.4M elems
  const size_t NNE = (size_t)BATCH * NTOK * NTOK;    // 16.8M elems
  const size_t WW  = (size_t)DIM * DIM;

  char* ws = (char*)d_ws;
  u16* adjb = (u16*)ws;                   ws += NNE * 2;   // adj bf16  (32 MiB)
  u16* buf0 = (u16*)ws;                   ws += ND * 2;    // x_hat -> t
  u16* buf1 = (u16*)ws;                   ws += ND * 2;    // h
  u16* in1b = (u16*)ws;                   ws += ND * 2;    // input_1 bf16
  u16* w0b  = (u16*)ws;                   ws += WW * 2;
  u16* w1b  = (u16*)ws;                   ws += WW * 2;
  u16* wfb  = (u16*)ws;                   ws += WW * 2;

  const long sND = (long)NTOK * DIM;
  const long sNN = (long)NTOK * NTOK;
  dim3 blk(256);
  dim3 gGram(NTOK / 64, NTOK / 128, BATCH);
  dim3 gProp(DIM / 64,  NTOK / 128, BATCH);

  // one-time bf16 conversions
  cvt_bf16_kernel<<<4096, blk, 0, stream>>>(input_1, in1b, ND);
  cvt_bf16_kernel<<<256,  blk, 0, stream>>>(w0, w0b, WW);
  cvt_bf16_kernel<<<256,  blk, 0, stream>>>(w1, w1b, WW);
  cvt_bf16_kernel<<<256,  blk, 0, stream>>>(wf, wfb, WW);

  // x_hat = l2norm(input_2) -> buf0 (bf16)
  l2norm_kernel<<<BATCH * NTOK, DIM, 0, stream>>>(input_2, buf0);

  // adj = mask(x_hat @ x_hat^T)  (NT, fused masking, bf16 out)
  gemm_bf16_wmma<true, false, true, false><<<gGram, blk, 0, stream>>>(
      buf0, buf0, nullptr, noun, adjb, DIM, DIM, DIM, NTOK, sND, sND, sNN);

  // t = adj @ input_1 (NN) -> buf0 (x_hat dead)
  gemm_bf16_wmma<false, false, false, false><<<gProp, blk, 0, stream>>>(
      adjb, in1b, nullptr, nullptr, buf0, NTOK, NTOK, DIM, DIM, sNN, sND, sND);
  // h1 = relu(t @ w0^T + b0) -> buf1
  gemm_bf16_wmma<true, true, false, false><<<gProp, blk, 0, stream>>>(
      buf0, w0b, b0, nullptr, buf1, DIM, DIM, DIM, DIM, sND, 0, sND);

  // t2 = adj @ h1 (NN) -> buf0
  gemm_bf16_wmma<false, false, false, false><<<gProp, blk, 0, stream>>>(
      adjb, buf1, nullptr, nullptr, buf0, NTOK, NTOK, DIM, DIM, sNN, sND, sND);
  // h2 = relu(t2 @ w1^T + b1) -> buf1
  gemm_bf16_wmma<true, true, false, false><<<gProp, blk, 0, stream>>>(
      buf0, w1b, b1, nullptr, buf1, DIM, DIM, DIM, DIM, sND, 0, sND);

  // out = relu(h2 @ wf^T + bf) -> d_out (f32)
  gemm_bf16_wmma<true, true, false, true><<<gProp, blk, 0, stream>>>(
      buf1, wfb, bf, nullptr, out, DIM, DIM, DIM, DIM, sND, 0, sND);
}
